// SetQFunction_40003325395311
// MI455X (gfx1250) — matrix operand order repaired and checked by reading, not compiled, not run on hardware
//
#include <hip/hip_runtime.h>

typedef __attribute__((ext_vector_type(16))) _Float16 v16h;
typedef __attribute__((ext_vector_type(8)))  float    v8f;
typedef unsigned int __attribute__((ext_vector_type(4))) v4u;
typedef int  __attribute__((ext_vector_type(8))) v8i;
typedef int  __attribute__((ext_vector_type(4))) v4i;

#define B_TOTAL 4096
#define N_OBST  128
#define HID     256
#define K1REAL  22
#define K1P     32
#define SPW     8            // samples per workgroup (encoder)
#define CK      416          // comb K padded (396 -> 416 = 13*32)
#define CKR     396
#define SPW2    64           // samples per workgroup (Q-head)

// ---- Encoder kernel LDS layout (bytes) ----
#define OFF_W1P   0          // 256*32 f16  = 16384
#define OFF_W2P   16384      // 256*256 f16 = 131072
#define OFF_X     147456     // 128*32 f16  = 8192
#define OFF_H1    155648     // 128*256 f16 = 65536
#define OFF_MASK  221184     // 128 f32     = 512
#define OFF_POOL  221696     // 256 f32     = 1024
#define SMEM1     222720

// ---- Q-head kernel LDS layout (bytes) ----
#define OFF_QWP   0          // max(13,8)*256*32 f16 = 212992 (qw1p) / 131072 (qw2p)
#define OFF_COMB2 212992     // 64*416 f16 = 53248
#define OFF_Q1B   266240     // 64*256 f16 = 32768
#define OFF_Q3    299008     // 64 f32     = 256
#define SMEM2     299264

union FragU { v16h v; uint4 u[2]; };

// Load a 16-half fragment as two 16B LDS chunks (ds_load_b128 x2).
static __device__ __forceinline__ v16h load_frag2(const _Float16* p0, const _Float16* p1) {
    FragU f;
    f.u[0] = *reinterpret_cast<const uint4*>(p0);
    f.u[1] = *reinterpret_cast<const uint4*>(p1);
    return f.v;
}

static __device__ __forceinline__ v8f wmma_f16(v16h a, v16h b, v8f c) {
    return __builtin_amdgcn_wmma_f32_16x16x32_f16(
        false, a, false, b, (short)0, c, false, false);
}

// =====================================================================
// Kernel 1: obstacle encoder + masked pooling; emits comb rows (f16) to ws
// =====================================================================
__global__ __launch_bounds__(256, 1)
void encoder_kernel(const float* __restrict__ obs,        // (B,138)
                    const float* __restrict__ obstacles,  // (B,5,128)
                    const float* __restrict__ act,        // (B,2)
                    const float* __restrict__ ow1,        // (22,256)
                    const float* __restrict__ ob1,        // (256)
                    const float* __restrict__ ow2,        // (256,256)
                    const float* __restrict__ ob2,        // (256)
                    _Float16* __restrict__ combg)         // (B,416) f16 out
{
    extern __shared__ __align__(16) char smem[];
    _Float16* w1p   = (_Float16*)(smem + OFF_W1P);
    _Float16* w2p   = (_Float16*)(smem + OFF_W2P);
    _Float16* xb    = (_Float16*)(smem + OFF_X);
    _Float16* h1    = (_Float16*)(smem + OFF_H1);
    float*    maskb = (float*)(smem + OFF_MASK);
    float*    pooled= (float*)(smem + OFF_POOL);

    const int tid  = threadIdx.x;
    const int lane = tid & 31;
    const int wv   = tid >> 5;
    const int l16  = lane & 15;
    const int kh   = lane >> 4;

    // Stage weights packed for B-fragments: wp[(kt*256 + n)*32 + k]
    for (int idx = tid; idx < K1P * HID; idx += 256) {
        int k = idx >> 8, n = idx & 255;
        float v = (k < K1REAL) ? ow1[k * HID + n] : 0.0f;
        w1p[n * K1P + k] = (_Float16)v;
    }
    for (int idx = tid; idx < HID * HID; idx += 256) {
        int kg = idx >> 8, n = idx & 255;
        int kt = kg >> 5, k = kg & 31;
        w2p[((kt << 8) + n) * 32 + k] = (_Float16)ow2[kg * HID + n];
    }
    __syncthreads();

    const int nb0 = (wv * 2) * 16;
    const int nb1 = (wv * 2 + 1) * 16;

    const _Float16* b1p0 = w1p + (nb0 + l16) * K1P + kh * 16;
    const _Float16* b1p1 = w1p + (nb1 + l16) * K1P + kh * 16;
    v16h bL1_0 = load_frag2(b1p0, b1p0 + 8);
    v16h bL1_1 = load_frag2(b1p1, b1p1 + 8);

    const float bias10 = ob1[nb0 + l16];
    const float bias11 = ob1[nb1 + l16];
    const float bias20 = ob2[nb0 + l16];
    const float bias21 = ob2[nb1 + l16];

    for (int s = 0; s < SPW; ++s) {
        const int b = blockIdx.x * SPW + s;

        for (int idx = tid; idx < N_OBST * K1P; idx += 256) {
            int r = idx >> 5, k = idx & 31;
            float v = 0.0f;
            if (k < 18)      v = obs[b * 138 + k];
            else if (k < 22) v = obstacles[b * 640 + (k - 18) * 128 + r];
            xb[idx] = (_Float16)v;
        }
        if (tid < 128) maskb[tid] = obstacles[b * 640 + 512 + tid];
        __syncthreads();

        // Layer 1: (128x32)@(32x256) + relu -> h1 (B frags held in VGPRs)
        for (int mt = 0; mt < 8; ++mt) {
            const int row = mt * 16 + l16;
            const _Float16* ap = xb + row * K1P + kh * 16;
            v16h a = load_frag2(ap, ap + 8);
            v8f c0 = {}; v8f c1 = {};
            c0 = wmma_f16(a, bL1_0, c0);
            c1 = wmma_f16(a, bL1_1, c1);
#pragma unroll
            for (int i = 0; i < 8; ++i) {
                int rr = mt * 16 + i + kh * 8;
                h1[rr * HID + nb0 + l16] = (_Float16)fmaxf(c0[i] + bias10, 0.0f);
                h1[rr * HID + nb1 + l16] = (_Float16)fmaxf(c1[i] + bias11, 0.0f);
            }
        }
        __syncthreads();

        // Layer 2: (128x256)@(256x256) + relu, fused mask-pool.
        // kt-outer / mt-inner: B fragments load once per kt; all 16
        // accumulator tiles (128 VGPRs) stay live across the K loop.
        v8f acc0[8], acc1[8];
#pragma unroll
        for (int mt = 0; mt < 8; ++mt) { acc0[mt] = {}; acc1[mt] = {}; }
        for (int kt = 0; kt < 8; ++kt) {
            const _Float16* bp0 = w2p + ((kt << 8) + nb0 + l16) * 32 + kh * 16;
            const _Float16* bp1 = w2p + ((kt << 8) + nb1 + l16) * 32 + kh * 16;
            v16h bb0 = load_frag2(bp0, bp0 + 8);
            v16h bb1 = load_frag2(bp1, bp1 + 8);
#pragma unroll
            for (int mt = 0; mt < 8; ++mt) {
                const _Float16* ap = h1 + (mt * 16 + l16) * HID + kt * 32 + kh * 16;
                v16h a = load_frag2(ap, ap + 8);
                acc0[mt] = wmma_f16(a, bb0, acc0[mt]);
                acc1[mt] = wmma_f16(a, bb1, acc1[mt]);
            }
        }
        float p0 = 0.0f, p1 = 0.0f;
#pragma unroll
        for (int mt = 0; mt < 8; ++mt) {
#pragma unroll
            for (int i = 0; i < 8; ++i) {
                int rr = mt * 16 + i + kh * 8;
                float mk = maskb[rr];
                p0 += mk * fmaxf(acc0[mt][i] + bias20, 0.0f);
                p1 += mk * fmaxf(acc1[mt][i] + bias21, 0.0f);
            }
        }
        p0 += __shfl_xor(p0, 16, 32);
        p1 += __shfl_xor(p1, 16, 32);
        if (lane < 16) {
            pooled[nb0 + lane] = p0;
            pooled[nb1 + lane] = p1;
        }
        __syncthreads();

        // Emit comb row = [obs(138) | pooled(256) | act(2) | zeros(20)] as f16
        for (int idx = tid; idx < CK; idx += 256) {
            float v = 0.0f;
            if (idx < 138)       v = obs[b * 138 + idx];
            else if (idx < 394)  v = pooled[idx - 138];
            else if (idx < 396)  v = act[b * 2 + (idx - 394)];
            combg[b * CK + idx] = (_Float16)v;
        }
        __syncthreads();
    }
}

// =====================================================================
// Kernel 2: batched Q-head, 64 samples per workgroup, all-WMMA.
// comb tile staged via the Tensor Data Mover (one instruction).
// =====================================================================
__global__ __launch_bounds__(256, 1)
void qhead_kernel(const _Float16* __restrict__ combg,  // (B,416) f16
                  const float* __restrict__ qw1,       // (396,256)
                  const float* __restrict__ qb1,       // (256)
                  const float* __restrict__ qw2,       // (256,256)
                  const float* __restrict__ qb2,       // (256)
                  const float* __restrict__ qw3,       // (256,1)
                  const float* __restrict__ qb3,       // (1)
                  float* __restrict__ out)             // (B)
{
    extern __shared__ __align__(16) char smem[];
    _Float16* qwp   = (_Float16*)(smem + OFF_QWP);
    _Float16* comb2 = (_Float16*)(smem + OFF_COMB2);
    _Float16* q1b   = (_Float16*)(smem + OFF_Q1B);
    float*    q3buf = (float*)(smem + OFF_Q3);

    const int tid  = threadIdx.x;
    const int lane = tid & 31;
    const int wv   = tid >> 5;
    const int l16  = lane & 15;
    const int kh   = lane >> 4;
    const int s0   = blockIdx.x * SPW2;

    // ---- TDM: DMA the 64x416 f16 comb tile into LDS (one instruction) ----
    if (wv == 0) {
        unsigned long long ga =
            (unsigned long long)(uintptr_t)(combg + (size_t)s0 * CK);
        unsigned int ldsa = (unsigned int)(uintptr_t)comb2;  // LDS byte addr (addr[31:0])

        v4u g0;
        g0.x = 1u;                                    // count=1, user-mode D#
        g0.y = ldsa;                                  // lds_addr
        g0.z = (unsigned int)(ga & 0xFFFFFFFFu);      // global_addr[31:0]
        g0.w = (unsigned int)((ga >> 32) & 0x1FFFFFFu)// global_addr[56:32]
             | (2u << 30);                            // type=2 ("image")

        v8i g1;
        g1[0] = (int)(1u << 16);                      // data_size=1 (2 bytes)
        g1[1] = (int)((CK & 0xFFFF) << 16);           // tensor_dim0[15:0]
        g1[2] = (int)((SPW2 & 0xFFFF) << 16);         // dim0[31:16]=0 | tensor_dim1[15:0]=64
        g1[3] = (int)(CK << 16);                      // dim1[31:16]=0 | tile_dim0=416
        g1[4] = (int)SPW2;                            // tile_dim1=64, tile_dim2=0
        g1[5] = (int)CK;                              // tensor_dim0_stride[31:0]=416
        g1[6] = 0;
        g1[7] = 0;

        v4i z4 = {0, 0, 0, 0};                        // groups 2/3 unused (2-D tensor)
        v8i z8 = {0, 0, 0, 0, 0, 0, 0, 0};
        __builtin_amdgcn_tensor_load_to_lds(g0, g1, z4, z4, z8, 0);
    }

    // Prefetch qw2 into GL2 while phase 1 runs (it is re-staged later).
    for (int idx = tid; idx < (HID * HID) / 32; idx += 256)
        __builtin_prefetch(qw2 + idx * 32, 0, 1);

    // Stage qw1 packed (K padded 396->416 with zeros) — needs f32->f16, VALU path.
    for (int idx = tid; idx < 13 * 32 * HID; idx += 256) {
        int kg = idx >> 8, n = idx & 255;
        int kt = kg >> 5, k = kg & 31;
        float v = (kg < CKR) ? qw1[kg * HID + n] : 0.0f;
        qwp[((kt << 8) + n) * 32 + k] = (_Float16)v;
    }
    if (tid < SPW2) q3buf[tid] = qb3[0];

    if (wv == 0) __builtin_amdgcn_s_wait_tensorcnt(0);  // comb tile landed
    __syncthreads();

    const int nb0 = (wv * 2) * 16;
    const int nb1 = (wv * 2 + 1) * 16;
    const float b10 = qb1[nb0 + l16];
    const float b11 = qb1[nb1 + l16];

    // q1 = relu(comb @ qw1 + qb1): kt-outer, 4 M-tiles held live
    {
        v8f c0[4], c1[4];
#pragma unroll
        for (int mt = 0; mt < 4; ++mt) { c0[mt] = {}; c1[mt] = {}; }
        for (int kt = 0; kt < 13; ++kt) {
            const _Float16* bp0 = qwp + ((kt << 8) + nb0 + l16) * 32 + kh * 16;
            const _Float16* bp1 = qwp + ((kt << 8) + nb1 + l16) * 32 + kh * 16;
            v16h bb0 = load_frag2(bp0, bp0 + 8);
            v16h bb1 = load_frag2(bp1, bp1 + 8);
#pragma unroll
            for (int mt = 0; mt < 4; ++mt) {
                const _Float16* ap = comb2 + (mt * 16 + l16) * CK + kt * 32 + kh * 16;
                v16h a = load_frag2(ap, ap + 8);
                c0[mt] = wmma_f16(a, bb0, c0[mt]);
                c1[mt] = wmma_f16(a, bb1, c1[mt]);
            }
        }
#pragma unroll
        for (int mt = 0; mt < 4; ++mt) {
#pragma unroll
            for (int i = 0; i < 8; ++i) {
                int rr = mt * 16 + i + kh * 8;
                q1b[rr * HID + nb0 + l16] = (_Float16)fmaxf(c0[mt][i] + b10, 0.0f);
                q1b[rr * HID + nb1 + l16] = (_Float16)fmaxf(c1[mt][i] + b11, 0.0f);
            }
        }
    }
    __syncthreads();

    // Re-stage same LDS region with qw2 packed (GL2-warm from the prefetch)
    for (int idx = tid; idx < HID * HID; idx += 256) {
        int kg = idx >> 8, n = idx & 255;
        int kt = kg >> 5, k = kg & 31;
        qwp[((kt << 8) + n) * 32 + k] = (_Float16)qw2[kg * HID + n];
    }
    __syncthreads();

    const float b20 = qb2[nb0 + l16];
    const float b21 = qb2[nb1 + l16];
    const float w30 = qw3[nb0 + l16];
    const float w31 = qw3[nb1 + l16];

    // q2 = relu(q1 @ qw2 + qb2); fold qw3 and row-reduce in-register
    {
        v8f c0[4], c1[4];
#pragma unroll
        for (int mt = 0; mt < 4; ++mt) { c0[mt] = {}; c1[mt] = {}; }
        for (int kt = 0; kt < 8; ++kt) {
            const _Float16* bp0 = qwp + ((kt << 8) + nb0 + l16) * 32 + kh * 16;
            const _Float16* bp1 = qwp + ((kt << 8) + nb1 + l16) * 32 + kh * 16;
            v16h bb0 = load_frag2(bp0, bp0 + 8);
            v16h bb1 = load_frag2(bp1, bp1 + 8);
#pragma unroll
            for (int mt = 0; mt < 4; ++mt) {
                const _Float16* ap = q1b + (mt * 16 + l16) * HID + kt * 32 + kh * 16;
                v16h a = load_frag2(ap, ap + 8);
                c0[mt] = wmma_f16(a, bb0, c0[mt]);
                c1[mt] = wmma_f16(a, bb1, c1[mt]);
            }
        }
#pragma unroll
        for (int mt = 0; mt < 4; ++mt) {
#pragma unroll
            for (int i = 0; i < 8; ++i) {
                float t = fmaxf(c0[mt][i] + b20, 0.0f) * w30
                        + fmaxf(c1[mt][i] + b21, 0.0f) * w31;
                // reduce the 16-lane column group (this wave covers 32 cols)
                t += __shfl_xor(t, 1, 32);
                t += __shfl_xor(t, 2, 32);
                t += __shfl_xor(t, 4, 32);
                t += __shfl_xor(t, 8, 32);
                if (l16 == 0) atomicAdd(&q3buf[mt * 16 + i + kh * 8], t);
            }
        }
    }
    __syncthreads();
    if (tid < SPW2) out[s0 + tid] = q3buf[tid];
}

extern "C" void kernel_launch(void* const* d_in, const int* in_sizes, int n_in,
                              void* d_out, int out_size, void* d_ws, size_t ws_size,
                              hipStream_t stream) {
    (void)in_sizes; (void)n_in; (void)out_size; (void)ws_size;
    const float* obs       = (const float*)d_in[0];
    const float* obstacles = (const float*)d_in[1];
    const float* act       = (const float*)d_in[2];
    const float* ow1       = (const float*)d_in[3];
    const float* ob1       = (const float*)d_in[4];
    const float* ow2       = (const float*)d_in[5];
    const float* ob2       = (const float*)d_in[6];
    const float* qw1       = (const float*)d_in[7];
    const float* qb1       = (const float*)d_in[8];
    const float* qw2       = (const float*)d_in[9];
    const float* qb2       = (const float*)d_in[10];
    const float* qw3       = (const float*)d_in[11];
    const float* qb3       = (const float*)d_in[12];
    float* out = (float*)d_out;
    _Float16* combg = (_Float16*)d_ws;   // 4096*416*2 = 3.4 MB scratch

    (void)hipFuncSetAttribute(reinterpret_cast<const void*>(encoder_kernel),
                              hipFuncAttributeMaxDynamicSharedMemorySize, SMEM1);
    (void)hipFuncSetAttribute(reinterpret_cast<const void*>(qhead_kernel),
                              hipFuncAttributeMaxDynamicSharedMemorySize, SMEM2);

    encoder_kernel<<<B_TOTAL / SPW, 256, SMEM1, stream>>>(
        obs, obstacles, act, ow1, ob1, ow2, ob2, combg);

    qhead_kernel<<<B_TOTAL / SPW2, 256, SMEM2, stream>>>(
        combg, qw1, qb1, qw2, qb2, qw3, qb3, out);
}
